// GaussianHistogram_55542517072143
// MI455X (gfx1250) — matrix operand order repaired and checked by reading, not compiled, not run on hardware
//
#include <hip/hip_runtime.h>
#include <hip/hip_bf16.h>

// ---------------------------------------------------------------------------
// GaussianHistogram on gfx1250 (MI455X):
//   hist[b] = G1[b] (256 x N) x G2[b]^T (N x 256),  COEF == 1.0 exactly.
// Fused: exp-tiles generated into LDS (f16, packed b32 stores), contracted
// with v_wmma_f32_16x16x32_f16 (8 chained WMMAs per wave per K-chunk),
// partials merged across N-slices with global_atomic_add_f32.
// 128x128 tiles -> only 2x exp recompute (exp is the roofline bottleneck).
// ---------------------------------------------------------------------------

typedef __attribute__((ext_vector_type(16))) _Float16 v16h;
typedef __attribute__((ext_vector_type(8)))  _Float16 v8h;
typedef __attribute__((ext_vector_type(2)))  _Float16 v2h;
typedef __attribute__((ext_vector_type(8)))  float    v8f;

#define BINS    256
#define MINV    (-0.25f)
#define DELTA   (1.5f / 256.0f)                          // (MAX-MIN)/BINS
#define INV_S   (1.0f / (DELTA * 0.3989422804014327f))   // 1/sigma
#define NTOT    32768
#define SLICE_N 2048
#define NSLICES (NTOT / SLICE_N)                // 16
#define CHUNK   32                              // WMMA K
#define TILE    128                             // 128x128 output tile per WG
#define JT      (TILE / 16)                     // 8 j-tiles per wave strip

static __device__ __forceinline__ v16h h8cat(v8h lo, v8h hi) {
    return __builtin_shufflevector(lo, hi, 0,1,2,3,4,5,6,7,8,9,10,11,12,13,14,15);
}

__global__ void ghist_zero_kernel(float* __restrict__ out, int nelem) {
    int i = blockIdx.x * blockDim.x + threadIdx.x;
    if (i < nelem) out[i] = 0.0f;
}

__global__ __launch_bounds__(256)
void ghist_wmma_kernel(const float* __restrict__ x1,
                       const float* __restrict__ x2,
                       const float* __restrict__ mask,
                       float* __restrict__ out)
{
    __shared__ alignas(16) _Float16 G1s[TILE][CHUNK];   // g1 rows = i-bins
    __shared__ alignas(16) _Float16 G2s[TILE][CHUNK];   // g2 rows = j-bins

    const int slice = blockIdx.x;           // 0..NSLICES-1
    const int ib    = blockIdx.y & 1;       // i block (128 bins)
    const int jb    = blockIdx.y >> 1;      // j block (128 bins)
    const int b     = blockIdx.z;           // batch

    const int tid  = threadIdx.x;
    const int lane = tid & 31;
    const int wave = tid >> 5;              // 0..7: wave owns 16x128 strip

    v8f acc[JT];
    #pragma unroll
    for (int t = 0; t < JT; ++t) acc[t] = (v8f){};

    // LDS fill mapping: thread owns a sample PAIR (packed b32 stores) and
    // row group rg, rg+16, ..., rg+112  -> 8 rows x 2 samples x 2 mats = 32 exps
    const int sp = (tid & 15) * 2;          // sample pair base: 0,2,..,30
    const int rg = tid >> 4;                // 0..15

    const float icen0 = MINV + DELTA * ((float)(ib * TILE) + 0.5f);
    const float jcen0 = MINV + DELTA * ((float)(jb * TILE) + 0.5f);

    const int base = b * NTOT + slice * SLICE_N;

    // Fragment addressing (ISA 7.12.2, wave32):
    //  A 16x32 f16: lane L -> row m=L&15; halves K = akb..+7 and 16+akb..+7
    const int am  = wave * 16 + (lane & 15);
    const int akb = (lane >> 4) * 8;
    //  B 32x16 f16: lane L -> col j=L&15; halves K = bkb..+7 and 16+bkb..+7
    const int bjl = lane & 15;
    const int bkb = (lane >> 4) * 8;

    for (int ch = 0; ch < SLICE_N; ch += CHUNK) {
        const float2 v1 = *(const float2*)&x1[base + ch + sp];
        const float2 v2 = *(const float2*)&x2[base + ch + sp];
        const float2 mk = *(const float2*)&mask[base + ch + sp];

        __syncthreads();    // previous iteration's fragment reads done
        #pragma unroll
        for (int r = 0; r < 8; ++r) {
            const int row = rg + r * 16;
            const float ic = icen0 + DELTA * (float)row;
            const float jc = jcen0 + DELTA * (float)row;
            const float d1a = (v1.x - ic) * INV_S;
            const float d1b = (v1.y - ic) * INV_S;
            const float d2a = (v2.x - jc) * INV_S;
            const float d2b = (v2.y - jc) * INV_S;
            v2h g1p, g2p;
            g1p.x = (_Float16)__expf(-0.5f * d1a * d1a);
            g1p.y = (_Float16)__expf(-0.5f * d1b * d1b);
            g2p.x = (_Float16)(__expf(-0.5f * d2a * d2a) * mk.x);
            g2p.y = (_Float16)(__expf(-0.5f * d2b * d2b) * mk.y);
            *(v2h*)&G1s[row][sp] = g1p;     // packed ds_store_b32
            *(v2h*)&G2s[row][sp] = g2p;
        }
        __syncthreads();

        // A fragment: loaded once, reused across 8 B tiles
        const v8h a_lo = *(const v8h*)&G1s[am][akb];
        const v8h a_hi = *(const v8h*)&G1s[am][16 + akb];
        const v16h af  = h8cat(a_lo, a_hi);

        #pragma unroll
        for (int t = 0; t < JT; ++t) {
            const int bj = t * 16 + bjl;
            const v8h b_lo = *(const v8h*)&G2s[bj][bkb];
            const v8h b_hi = *(const v8h*)&G2s[bj][bkb + 16];
            const v16h bf  = h8cat(b_lo, b_hi);
            acc[t] = __builtin_amdgcn_wmma_f32_16x16x32_f16(
                         false, af, false, bf, (short)0, acc[t], false, false);
        }
    }

    // C/D layout: VGPR r, lane L -> row = r + 8*(L>>4), col = L&15
    const int om  = ib * TILE + wave * 16 + (lane >> 4) * 8;
    const int on0 = jb * TILE + (lane & 15);
    float* outb = out + (size_t)b * BINS * BINS;

    #pragma unroll
    for (int t = 0; t < JT; ++t) {
        #pragma unroll
        for (int r = 0; r < 8; ++r) {
            unsafeAtomicAdd(&outb[(om + r) * BINS + on0 + t * 16], acc[t][r]);
        }
    }
}

extern "C" void kernel_launch(void* const* d_in, const int* in_sizes, int n_in,
                              void* d_out, int out_size, void* d_ws, size_t ws_size,
                              hipStream_t stream) {
    (void)in_sizes; (void)n_in; (void)d_ws; (void)ws_size;
    const float* x1   = (const float*)d_in[0];
    const float* x2   = (const float*)d_in[1];
    const float* mask = (const float*)d_in[2];
    float* out = (float*)d_out;

    // d_out is poisoned by the harness; zero it (we accumulate atomically).
    ghist_zero_kernel<<<(out_size + 255) / 256, 256, 0, stream>>>(out, out_size);

    dim3 grid(NSLICES, 4, 8);   // x: N-slice, y: 2x2 output blocks, z: batch
    ghist_wmma_kernel<<<grid, 256, 0, stream>>>(x1, x2, mask, out);
}
